// CompletePatchReadout_61942018343194
// MI455X (gfx1250) — matrix operand order repaired and checked by reading, not compiled, not run on hardware
//
#include <hip/hip_runtime.h>

// Problem constants (from reference)
constexpr int B    = 64;
constexpr int T    = 12;
constexpr int P    = 128;
constexpr int F    = 128;
constexpr int H    = 12;
constexpr int MAXC = 48;
constexpr int D    = T * F;      // 1536 (K dimension)
constexpr int NF   = MAXC * H;   // 576  (N dimension)
constexpr int KC   = 32;         // K per bf16 WMMA step
constexpr int MTILES  = 4;       // 64 batch rows = 4 x 16
constexpr int WAVES   = 6;       // one 16-wide N-tile per wave
constexpr int THREADS = WAVES * 32;
constexpr int ROWB    = 80;      // LDS row stride in bytes (64B data + 16B pad)

typedef __attribute__((ext_vector_type(16))) __bf16 v16bf;
typedef __attribute__((ext_vector_type(8)))  __bf16 v8bf;
typedef __attribute__((ext_vector_type(4)))  __bf16 v4bf;
typedef __attribute__((ext_vector_type(8)))  float  v8f;

__global__ __launch_bounds__(THREADS)
void patch_readout_wmma(const float* __restrict__ x,
                        const float* __restrict__ W,
                        const float* __restrict__ bias,
                        const int*   __restrict__ node_map,
                        float*       __restrict__ out,
                        int NN)
{
    // A-chunk staging buffer: 64 rows x 32 bf16 (64B), padded stride
    __shared__ __align__(16) unsigned char smem[64 * ROWB];

    const int p    = blockIdx.y;          // patch
    const int tid  = threadIdx.x;
    const int wave = tid >> 5;            // wave32
    const int lane = tid & 31;
    const int half = lane >> 4;           // 0: lanes 0-15, 1: lanes 16-31
    const int lr   = lane & 15;

    const int jt = blockIdx.x * WAVES + wave;   // N-tile 0..35
    const int n0 = jt * 16;

    const float* __restrict__ Wp = W + (size_t)p * D * NF;
    const int wcol = n0 + lr;             // this lane's N column in W

    v8f acc[MTILES] = {};

    for (int k0 = 0; k0 < D; k0 += KC) {
        const int t  = k0 >> 7;           // k0 / F  (KC divides F, so t fixed per chunk)
        const int f0 = k0 & (F - 1);

        // ---- W fragment: per-lane column of 16 f32 (K = k0+16*half .. +15), stride NF ----
        const float* wb = Wp + (size_t)(k0 + half * 16) * NF + wcol;
        float wv[16];
        #pragma unroll
        for (int j = 0; j < 16; ++j) wv[j] = wb[(size_t)j * NF];

        // ---- stage A chunk (64 batch rows x 32 K-values) into LDS as bf16 ----
        __syncthreads();                  // previous chunk's reads complete
        for (int i = tid; i < 64 * 8; i += THREADS) {
            const int row = i >> 3;       // batch index 0..63
            const int seg = i & 7;        // 4-float segment within 32 K
            const float4 v = *(const float4*)(
                x + (((size_t)row * T + t) * P + p) * F + f0 + seg * 4);
            v4bf pk = { (__bf16)v.x, (__bf16)v.y, (__bf16)v.z, (__bf16)v.w };
            *(v4bf*)(smem + row * ROWB + seg * 8) = pk;
        }
        __syncthreads();

        // ---- B fragment (32x16 K-major, contiguous k = 16*half + j, matching A) ----
        union { v16bf v; __bf16 e[16]; } bf;
        #pragma unroll
        for (int j = 0; j < 16; ++j) bf.e[j] = (__bf16)wv[j];

        // ---- 4 M-tiles x WMMA f32 = bf16 x bf16 + f32 ----
        #pragma unroll
        for (int mt = 0; mt < MTILES; ++mt) {
            const unsigned char* rb = smem + (mt * 16 + lr) * ROWB;
            union { v16bf v; v8bf h[2]; } af;
            // elements 0..7 : K = 16*half .. 16*half+7
            af.h[0] = *(const v8bf*)(rb + half * 32);
            // elements 8..15: K = 16*half + 8 .. +7
            af.h[1] = *(const v8bf*)(rb + half * 32 + 16);
            acc[mt] = __builtin_amdgcn_wmma_f32_16x16x32_bf16(
                false, af.v, false, bf.v, (short)0, acc[mt], false, false);
        }
    }

    // ---- epilogue: bias add + node_map scatter ----
    const int nflat = n0 + lr;            // this lane's flat N index (fixed per lane)
    const int nc    = nflat / H;
    const int h     = nflat - nc * H;
    const int node  = node_map[p * MAXC + nc];
    const float bv  = bias[(size_t)p * NF + nflat];
    if (node < NN) {
        #pragma unroll
        for (int mt = 0; mt < MTILES; ++mt) {
            #pragma unroll
            for (int r = 0; r < 8; ++r) {
                const int bch = mt * 16 + half * 8 + r;   // C layout: M = r + 8*half
                out[((size_t)node * B + bch) * H + h] = acc[mt][r] + bv;
            }
        }
    }
}

extern "C" void kernel_launch(void* const* d_in, const int* in_sizes, int n_in,
                              void* d_out, int out_size, void* d_ws, size_t ws_size,
                              hipStream_t stream) {
    const float* x  = (const float*)d_in[0];
    const float* W  = (const float*)d_in[1];
    const float* b  = (const float*)d_in[2];
    const int*   nm = (const int*)d_in[3];
    float* out = (float*)d_out;
    const int NN = out_size / (B * H);     // number of valid nodes
    dim3 grid(36 / WAVES, P);              // 6 N-groups x 128 patches
    patch_readout_wmma<<<grid, THREADS, 0, stream>>>(x, W, b, nm, out, NN);
}
